// set_net_39058432590153
// MI455X (gfx1250) — compile-verified, matrix operand order checked by reading, not run on hardware
//
#include <hip/hip_runtime.h>
#include <math.h>

// ---- CDNA5 WMMA / TDM types ----
typedef __attribute__((ext_vector_type(16))) _Float16     v16h;
typedef __attribute__((ext_vector_type(8)))  float        v8f;
typedef __attribute__((ext_vector_type(4)))  unsigned int v4u;
typedef __attribute__((ext_vector_type(8)))  int          v8i;
typedef __attribute__((ext_vector_type(4)))  int          v4i;

#define B_SETS 256
#define S_SET  1024
#define D_IN   128
#define H_DIM  256

// ---------------------------------------------------------------------------
// Convert left-half of a [256, 2d] f32 weight matrix to dense [256, d] f16.
// ---------------------------------------------------------------------------
__global__ void cvt_w_left(const float* __restrict__ W, _Float16* __restrict__ Wl, int d) {
    int idx = blockIdx.x * blockDim.x + threadIdx.x;
    if (idx >= H_DIM * d) return;
    int row = idx / d, k = idx - row * d;
    Wl[idx] = (_Float16)W[(size_t)row * (2 * d) + k];
}

// ---------------------------------------------------------------------------
// Per-set column mean: out[b, c] = scale * sum_s Y[b, s, c].
// ---------------------------------------------------------------------------
template <typename T>
__global__ void mean_over_set(const T* __restrict__ Y, float* __restrict__ out,
                              int d, float scale) {
    const int b = blockIdx.x;
    const int c = threadIdx.x;
    const T* p = Y + (size_t)b * S_SET * d + c;
    float s = 0.f;
    #pragma unroll 4
    for (int i = 0; i < S_SET; ++i) s += (float)p[(size_t)i * d];
    out[(size_t)b * d + c] = s * scale;
}

// ---------------------------------------------------------------------------
// Effective bias: beff[b,h] = bias[h] + sum_k W[h, d+k] * m[b,k]
// ---------------------------------------------------------------------------
__global__ void bias_eff(const float* __restrict__ W,    // [256, 2d] f32
                         const float* __restrict__ bvec, // [256]
                         const float* __restrict__ m,    // [B, d]
                         float* __restrict__ beff,       // [B, 256]
                         int d) {
    const int b = blockIdx.x;
    const int h = threadIdx.x;
    const float* wr = W + (size_t)h * (2 * d) + d;
    const float* mb = m + (size_t)b * d;
    float acc = bvec[h];
    #pragma unroll 4
    for (int k = 0; k < d; ++k) acc += wr[k] * mb[k];
    beff[(size_t)b * H_DIM + h] = acc;
}

// ---------------------------------------------------------------------------
// Y[b] = tanh(A[b] @ Wl^T + beff[b]) with f16 WMMA, f32 accumulate.
// Block tile 64(M) x 256(N), 8 waves as 4M x 2N, wave tile 16 x 128.
// Weights staged into LDS in 128-wide K-chunks by the Tensor Data Mover
// (one tensor_load_to_lds per chunk, TENSORcnt-tracked), WMMA B-operands
// read back with ds_load_b128.
// ---------------------------------------------------------------------------
template <typename AT, int K>
__global__ __launch_bounds__(256) void gemm_bias_tanh(
    const AT* __restrict__ A,
    const _Float16* __restrict__ Wl,
    const float* __restrict__ beff,
    _Float16* __restrict__ Y) {
    constexpr int KC = 128;                 // K-chunk resident in LDS
    __shared__ _Float16 lds[H_DIM * KC];    // 64 KB

    const int batch  = blockIdx.y;
    const int tid    = threadIdx.x;
    const int lane   = tid & 31;
    const int wid    = tid >> 5;
    const int waveM  = wid >> 1;            // 0..3
    const int waveN  = wid & 1;             // 0..1
    const int rowBase = blockIdx.x * 64 + waveM * 16;
    const int nBase   = waveN * 128;
    const int mlane   = lane & 15;
    const int khalf   = (lane >> 4) * 8;

    const AT* aRow = A + ((size_t)batch * S_SET + rowBase + mlane) * K + khalf;

    v8f zero = {};
    v8f acc[8];
    #pragma unroll
    for (int t = 0; t < 8; ++t) acc[t] = zero;

    for (int kc = 0; kc < K; kc += KC) {
        // Prefetch next chunk of A rows while weights are staged.
        if (kc + KC < K) __builtin_prefetch(aRow + kc + KC, 0, 0);

        __syncthreads();  // previous chunk's LDS readers done (WAR)

#if __has_builtin(__builtin_amdgcn_tensor_load_to_lds)
        if (wid == 0) {
            // ---- Tensor DMA descriptor (ISA 8.3-8.6), 2-D tile ----
            // Tile: KC cols x H_DIM rows of 2-byte elements,
            // global row stride = K elements, LDS dest contiguous.
            const unsigned long long ga = (unsigned long long)(uintptr_t)(Wl + kc);
            // Flat shared address low 32 bits == LDS byte offset (aperture map).
            const unsigned int lds_off = (unsigned int)(uintptr_t)(void*)lds;
            v4u g0 = {
                1u,                                    // count=1, user mode
                lds_off,                               // lds_addr
                (unsigned int)(ga & 0xffffffffu),      // global_addr[31:0]
                (unsigned int)((ga >> 32) & 0x01ffffffu) | (2u << 30) // ga[56:32]|type=2
            };
            v8i g1 = {
                (int)(1u << 16),                         // wg_mask=0, data_size=1 (2B)
                (int)((unsigned)KC << 16),               // tensor_dim0[15:0]=KC
                (int)(((unsigned)H_DIM & 0xffffu) << 16),// td0 hi=0 | tensor_dim1 lo=256
                (int)((unsigned)KC << 16),               // td1 hi=0 | tile_dim0=KC
                (int)H_DIM,                              // tile_dim1=256, tile_dim2=0
                (int)K,                                  // tensor_dim0_stride lo = K
                0,                                       // td0_stride hi | td1_stride lo
                0                                        // tensor_dim1_stride hi
            };
            v4i gz = { 0, 0, 0, 0 };
#if defined(__clang_major__) && (__clang_major__ >= 23)
            v8i gz8 = { 0, 0, 0, 0, 0, 0, 0, 0 };
            __builtin_amdgcn_tensor_load_to_lds(g0, g1, gz, gz, gz8, 0);
#else
            __builtin_amdgcn_tensor_load_to_lds(g0, g1, gz, gz, 0);
#endif
            __builtin_amdgcn_s_wait_tensorcnt(0);
        }
        // Compiler must assume LDS was written by the TDM op.
        asm volatile("" : : "r"(lds) : "memory");
#else
        {   // Fallback: manual 16B-unit copy into LDS.
            constexpr int unitsPerRow = KC * 2 / 16;
            constexpr int totalUnits  = H_DIM * unitsPerRow;
            #pragma unroll 4
            for (int u = tid; u < totalUnits; u += 256) {
                const int row = u / unitsPerRow;
                const int off = u % unitsPerRow;
                ((uint4*)(lds + row * KC))[off] =
                    ((const uint4*)(Wl + (size_t)row * K + kc))[off];
            }
        }
#endif
        __syncthreads();  // staged chunk visible to all waves

        #pragma unroll
        for (int k0 = 0; k0 < KC; k0 += 32) {
            // A fragment from global (converted in-register for f32 layer 0)
            v16h afrag;
            {
                const AT* p = aRow + kc + k0;
                #pragma unroll
                for (int i = 0; i < 8; ++i) {
                    afrag[i]     = (_Float16)p[i];
                    afrag[i + 8] = (_Float16)p[i + 16];
                }
            }
            #pragma unroll
            for (int t = 0; t < 8; ++t) {
                const _Float16* wp =
                    lds + (nBase + t * 16 + mlane) * KC + k0 + khalf;
                v16h bfrag;   // two ds_load_b128
                #pragma unroll
                for (int i = 0; i < 8; ++i) {
                    bfrag[i]     = wp[i];
                    bfrag[i + 8] = wp[i + 16];
                }
                acc[t] = __builtin_amdgcn_wmma_f32_16x16x32_f16(
                    /*neg_a=*/false, afrag, /*neg_b=*/false, bfrag,
                    /*c_mod=*/(short)0, acc[t], /*reuse_a=*/false, /*reuse_b=*/false);
            }
        }
    }

    // Epilogue: C/D layout -> VGPR r: lanes 0-15 = row r, lanes 16-31 = row r+8
    const int rowOff = (lane >> 4) * 8;
    #pragma unroll
    for (int t = 0; t < 8; ++t) {
        const int col = nBase + t * 16 + mlane;
        const float bb = beff[(size_t)batch * H_DIM + col];
        #pragma unroll
        for (int r = 0; r < 8; ++r) {
            const int row = rowBase + r + rowOff;
            const float v = tanhf(acc[t][r] + bb);
            Y[((size_t)batch * S_SET + row) * H_DIM + col] = (_Float16)v;
        }
    }
}

// ---------------------------------------------------------------------------
extern "C" void kernel_launch(void* const* d_in, const int* in_sizes, int n_in,
                              void* d_out, int out_size, void* d_ws, size_t ws_size,
                              hipStream_t stream) {
    const float* x  = (const float*)d_in[0];
    const float* W0 = (const float*)d_in[1];
    const float* b0 = (const float*)d_in[2];
    const float* W1 = (const float*)d_in[3];
    const float* b1 = (const float*)d_in[4];
    const float* W2 = (const float*)d_in[5];
    const float* b2 = (const float*)d_in[6];
    float* out = (float*)d_out;

    // Workspace layout
    char* ws = (char*)d_ws;
    const size_t szY = (size_t)B_SETS * S_SET * H_DIM * sizeof(_Float16); // 134 MB
    _Float16* Y0 = (_Float16*)(ws);
    _Float16* Y1 = (_Float16*)(ws + szY);
    char* p = ws + 2 * szY;
    _Float16* W0L = (_Float16*)p; p += (size_t)H_DIM * D_IN  * sizeof(_Float16);
    _Float16* W1L = (_Float16*)p; p += (size_t)H_DIM * H_DIM * sizeof(_Float16);
    _Float16* W2L = (_Float16*)p; p += (size_t)H_DIM * H_DIM * sizeof(_Float16);
    float* mbuf = (float*)p;      p += (size_t)B_SETS * H_DIM * sizeof(float);
    float* beff = (float*)p;

    const float invS = 1.0f / (float)S_SET;
    const dim3 gGemm(S_SET / 64, B_SETS);

    // Weight left-half conversion to f16
    cvt_w_left<<<(H_DIM * D_IN  + 255) / 256, 256, 0, stream>>>(W0, W0L, D_IN);
    cvt_w_left<<<(H_DIM * H_DIM + 255) / 256, 256, 0, stream>>>(W1, W1L, H_DIM);
    cvt_w_left<<<(H_DIM * H_DIM + 255) / 256, 256, 0, stream>>>(W2, W2L, H_DIM);

    // Layer 0: K = 128, A = x (f32, converted in-register)
    mean_over_set<float><<<B_SETS, D_IN, 0, stream>>>(x, mbuf, D_IN, invS);
    bias_eff<<<B_SETS, H_DIM, 0, stream>>>(W0, b0, mbuf, beff, D_IN);
    gemm_bias_tanh<float, D_IN><<<gGemm, 256, 0, stream>>>(x, W0L, beff, Y0);

    // Layer 1: K = 256, A = Y0 (f16)
    mean_over_set<_Float16><<<B_SETS, H_DIM, 0, stream>>>(Y0, mbuf, H_DIM, invS);
    bias_eff<<<B_SETS, H_DIM, 0, stream>>>(W1, b1, mbuf, beff, H_DIM);
    gemm_bias_tanh<_Float16, H_DIM><<<gGemm, 256, 0, stream>>>(Y0, W1L, beff, Y1);

    // Layer 2: K = 256, A = Y1 (f16), write back into Y0
    mean_over_set<_Float16><<<B_SETS, H_DIM, 0, stream>>>(Y1, mbuf, H_DIM, invS);
    bias_eff<<<B_SETS, H_DIM, 0, stream>>>(W2, b2, mbuf, beff, H_DIM);
    gemm_bias_tanh<_Float16, H_DIM><<<gGemm, 256, 0, stream>>>(Y1, W2L, beff, Y0);

    // Final mean over the set -> f32 output [B, 256]
    mean_over_set<_Float16><<<B_SETS, H_DIM, 0, stream>>>(Y0, out, H_DIM, invS);
}
